// DenseLtownLocGcn_33097017983380
// MI455X (gfx1250) — matrix operand order repaired and checked by reading, not compiled, not run on hardware
//
#include <hip/hip_runtime.h>

// ---------------------------------------------------------------------------
// CDNA5 (gfx1250, wave32) implementation of DenseLtownLocGcn forward.
// All dense math runs through v_wmma_f32_16x16x32_bf16 (bf16 in, f32 acc).
// GEMMs use ping-pong LDS tiles (1 barrier / k-step) + vectorized staging.
// ---------------------------------------------------------------------------

typedef unsigned short u16;
typedef unsigned int   u32;
typedef __bf16 bf16;
typedef bf16  v16bf __attribute__((ext_vector_type(16)));
typedef float v8f   __attribute__((ext_vector_type(8)));
typedef u32   uv4   __attribute__((ext_vector_type(4)));
typedef u32   uv2   __attribute__((ext_vector_type(2)));

__device__ __forceinline__ u16 f2b(float f) {           // fp32 -> bf16 (RNE)
  u32 u = __float_as_uint(f);
  u32 r = (u + 0x7FFFu + ((u >> 16) & 1u)) >> 16;
  return (u16)r;
}
__device__ __forceinline__ u32 f2ord(float f) {          // order-preserving uint
  u32 u = __float_as_uint(f);
  return (u & 0x80000000u) ? ~u : (u | 0x80000000u);
}
__device__ __forceinline__ float ord2f(u32 o) {
  return (o & 0x80000000u) ? __uint_as_float(o & 0x7FFFFFFFu)
                           : __uint_as_float(~o);
}

union FragAB { v16bf v; uv4 q[2]; };
union FragC  { v8f   v; float f[8]; };

// ---------------- shared GEMM tile helpers (tile 128x64, K-step 32) --------
// A: [M, lda] bf16 row-major, lda % 8 == 0 (16B aligned rows).
// B: [K, ldb] bf16 row-major, ldb % 4 == 0 (8B vector loads), N % 4 == 0.
// LDS rows padded to 40 u16 (80B) -> conflict-free 16-lane fragment reads.

__device__ __forceinline__ void load_a_tile(const u16* __restrict__ A, int lda,
                                            int M, int bm, int rowOff, int k0,
                                            int tid, uv4 aR[2]) {
#pragma unroll
  for (int i = 0; i < 2; ++i) {
    int q = tid + i * 256, m = q >> 2, c8 = (q & 3) << 3;
    uv4 v = {0u, 0u, 0u, 0u};
    int gm = bm + m + rowOff;
    if (gm >= 0 && gm < M) v = *(const uv4*)(A + (size_t)gm * lda + k0 + c8);
    aR[i] = v;
  }
}
__device__ __forceinline__ void load_b_tile(const u16* __restrict__ B, int ldb,
                                            int N, int bn, int krow,
                                            int tid, uv2 bR[2]) {
#pragma unroll
  for (int i = 0; i < 2; ++i) {
    int q = tid + i * 256, kk = q >> 4, n4 = (q & 15) << 2;
    uv2 v = {0u, 0u};
    int gn = bn + n4;
    if (gn < N) v = *(const uv2*)(B + (size_t)(krow + kk) * ldb + gn);
    bR[i] = v;
  }
}
__device__ __forceinline__ void store_tiles(u16 (*As)[40], u16 (*Bt)[40],
                                            int tid, const uv4 aR[2], const uv2 bR[2]) {
#pragma unroll
  for (int i = 0; i < 2; ++i) {
    int q = tid + i * 256, m = q >> 2, c8 = (q & 3) << 3;
    *(uv4*)&As[m][c8] = aR[i];
  }
#pragma unroll
  for (int i = 0; i < 2; ++i) {                 // transpose B into Bt[n][k]
    int q = tid + i * 256, kk = q >> 4, n4 = (q & 15) << 2;
    u32 w0 = bR[i].x, w1 = bR[i].y;
    Bt[n4 + 0][kk] = (u16)w0;
    Bt[n4 + 1][kk] = (u16)(w0 >> 16);
    Bt[n4 + 2][kk] = (u16)w1;
    Bt[n4 + 3][kk] = (u16)(w1 >> 16);
  }
}
__device__ __forceinline__ void compute_tile(const u16 (*As)[40], const u16 (*Bt)[40],
                                             int wm, int wn, int half, int l16,
                                             FragC acc[2][2]) {
  FragAB a[2], b[2];
#pragma unroll
  for (int i = 0; i < 2; ++i) {
    const u16* ap = &As[wm + i * 16 + l16][half * 8];   // K = half*8..+7, 16+half*8..+7
    a[i].q[0] = *(const uv4*)ap;
    a[i].q[1] = *(const uv4*)(ap + 16);
    const u16* bp = &Bt[wn + i * 16 + l16][half * 8];   // K = half*8..+7, 16+half*8..+7
    b[i].q[0] = *(const uv4*)bp;
    b[i].q[1] = *(const uv4*)(bp + 16);
  }
#pragma unroll
  for (int i = 0; i < 2; ++i)
#pragma unroll
    for (int j = 0; j < 2; ++j)
      acc[i][j].v = __builtin_amdgcn_wmma_f32_16x16x32_bf16(
          false, a[i].v, false, b[j].v, (short)0, acc[i][j].v, false, false);
}

// ---------------------------------------------------------------------------
// Generic GEMM: C[M,N] = A[M,K] * B[K,N] + bias, opt ReLU.  K % 32 == 0.
// Epilogue writes fp32 (ldcf) and/or zero-padded bf16 (ldcb).
// ---------------------------------------------------------------------------
__global__ __launch_bounds__(256) void k_gemm(
    const u16* __restrict__ A, int lda,
    const u16* __restrict__ B, int ldb,
    const float* __restrict__ bias,
    float* __restrict__ Cf, int ldcf,
    u16* __restrict__ Cb, int ldcb,
    int M, int N, int K, int relu)
{
  __shared__ u16 As[2][128][40];
  __shared__ u16 Bt[2][64][40];
  const int tid  = threadIdx.x;
  const int bm   = blockIdx.y * 128, bn = blockIdx.x * 64;
  const int lane = tid & 31, wid = tid >> 5;
  const int wm   = (wid & 3) * 32, wn = (wid >> 2) * 32;
  const int half = lane >> 4, l16 = lane & 15;

  FragC acc[2][2];
#pragma unroll
  for (int i = 0; i < 2; ++i)
#pragma unroll
    for (int j = 0; j < 2; ++j)
#pragma unroll
      for (int r = 0; r < 8; ++r) acc[i][j].f[r] = 0.0f;

  const int nK = K >> 5;
  uv4 aR[2]; uv2 bR[2];
  load_a_tile(A, lda, M, bm, 0, 0, tid, aR);
  load_b_tile(B, ldb, N, bn, 0, tid, bR);
  store_tiles(As[0], Bt[0], tid, aR, bR);
  __syncthreads();

  int cur = 0;
  for (int kc = 0; kc < nK; ++kc) {
    if (kc + 1 < nK) {                           // prefetch next K-chunk
      load_a_tile(A, lda, M, bm, 0, (kc + 1) << 5, tid, aR);
      load_b_tile(B, ldb, N, bn, (kc + 1) << 5, tid, bR);
    }
    compute_tile(As[cur], Bt[cur], wm, wn, half, l16, acc);
    if (kc + 1 < nK) {
      cur ^= 1;
      store_tiles(As[cur], Bt[cur], tid, aR, bR);
      __syncthreads();
    }
  }

#pragma unroll
  for (int i = 0; i < 2; ++i) {
    int rowb = bm + wm + i * 16 + half * 8;
#pragma unroll
    for (int j = 0; j < 2; ++j) {
      int colC = bn + wn + j * 16 + l16;
#pragma unroll
      for (int r = 0; r < 8; ++r) {
        int row = rowb + r;
        if (row >= M) continue;
        float v = acc[i][j].f[r];
        if (colC < N) {
          if (bias) v += bias[colC];
          if (relu) v = fmaxf(v, 0.0f);
          if (Cf) Cf[(size_t)row * ldcf + colC] = v;
          if (Cb) Cb[(size_t)row * ldcb + colC] = f2b(v);
        } else if (Cb && colC < ldcb) {
          Cb[(size_t)row * ldcb + colC] = 0;      // zero the K-padding columns
        }
      }
    }
  }
}

// ---------------------------------------------------------------------------
// Fused multi-scale Conv1d bank as a shifted-A GEMM:
// out[m, :] = sum_t X[m + t - 9, :cinP] * Wbig[t*cinP : (t+1)*cinP, :] + bias
// (zero rows outside [0,M)).  Emits bf16 output with padded columns.
// ---------------------------------------------------------------------------
__global__ __launch_bounds__(256) void k_convgemm(
    const u16* __restrict__ X, int cinP,
    const u16* __restrict__ B,          // Wbig [19*cinP, NO] row-major
    const float* __restrict__ bias,
    u16* __restrict__ Cb, int ldcb,
    int M, int NO)
{
  __shared__ u16 As[2][128][40];
  __shared__ u16 Bt[2][64][40];
  const int tid  = threadIdx.x;
  const int bm   = blockIdx.y * 128, bn = blockIdx.x * 64;
  const int lane = tid & 31, wid = tid >> 5;
  const int wm   = (wid & 3) * 32, wn = (wid >> 2) * 32;
  const int half = lane >> 4, l16 = lane & 15;

  FragC acc[2][2];
#pragma unroll
  for (int i = 0; i < 2; ++i)
#pragma unroll
    for (int j = 0; j < 2; ++j)
#pragma unroll
      for (int r = 0; r < 8; ++r) acc[i][j].f[r] = 0.0f;

  const int KC = cinP >> 5;
  const int nK = 19 * KC;
  uv4 aR[2]; uv2 bR[2];
  load_a_tile(X, cinP, M, bm, -9, 0, tid, aR);   // kIdx 0: t=0, k0=0
  load_b_tile(B, NO, NO, bn, 0, tid, bR);
  store_tiles(As[0], Bt[0], tid, aR, bR);
  __syncthreads();

  int cur = 0;
  for (int kIdx = 0; kIdx < nK; ++kIdx) {
    if (kIdx + 1 < nK) {                         // prefetch next (t, k0) chunk
      int nx = kIdx + 1;
      int t  = nx / KC;
      int k0 = (nx - t * KC) << 5;
      load_a_tile(X, cinP, M, bm, t - 9, k0, tid, aR);
      load_b_tile(B, NO, NO, bn, nx << 5, tid, bR);
    }
    compute_tile(As[cur], Bt[cur], wm, wn, half, l16, acc);
    if (kIdx + 1 < nK) {
      cur ^= 1;
      store_tiles(As[cur], Bt[cur], tid, aR, bR);
      __syncthreads();
    }
  }

#pragma unroll
  for (int i = 0; i < 2; ++i) {
    int rowb = bm + wm + i * 16 + half * 8;
#pragma unroll
    for (int j = 0; j < 2; ++j) {
      int colC = bn + wn + j * 16 + l16;
#pragma unroll
      for (int r = 0; r < 8; ++r) {
        int row = rowb + r;
        if (row >= M) continue;
        if (colC < NO) {
          float v = acc[i][j].f[r] + bias[colC];
          Cb[(size_t)row * ldcb + colC] = f2b(v);
        } else if (colC < ldcb) {
          Cb[(size_t)row * ldcb + colC] = 0;
        }
      }
    }
  }
}

// ------------------------- small utility kernels ---------------------------
struct ConvW { const float* w[10]; const float* b[10]; };

__global__ void k_build_wbig(ConvW cw, u16* wbig, int cin, int cinP, int cout) {
  int NO = 10 * cout;
  long long idx = (long long)blockIdx.x * 256 + threadIdx.x;
  long long total = (long long)19 * cinP * NO;
  if (idx >= total) return;
  int o = (int)(idx % NO);
  long long kk = idx / NO;
  int t = (int)(kk / cinP), c = (int)(kk % cinP);
  float v = 0.0f;
  if (c < cin) {
    int i = o / cout, oc = o % cout;
    int ks = 2 * i + 1, tap = t - (9 - i);
    if (tap >= 0 && tap < ks) v = cw.w[i][((size_t)oc * cin + c) * ks + tap];
  }
  wbig[idx] = f2b(v);
}
__global__ void k_bbig(ConvW cw, float* bbig, int cout) {
  int o = blockIdx.x * 256 + threadIdx.x;
  if (o < 10 * cout) bbig[o] = cw.b[o / cout][o % cout];
}
__global__ void k_zero_f32(float* p, long long n) {
  long long i = (long long)blockIdx.x * 256 + threadIdx.x;
  if (i < n) p[i] = 0.0f;
}
__global__ void k_zero_u32(u32* p, long long n) {
  long long i = (long long)blockIdx.x * 256 + threadIdx.x;
  if (i < n) p[i] = 0u;
}
__global__ void k_cvt_pad(const float* src, u16* dst, int M, int cs, int cp) {
  long long i = (long long)blockIdx.x * 256 + threadIdx.x;
  if (i >= (long long)M * cp) return;
  int m = (int)(i / cp), c = (int)(i % cp);
  dst[i] = (c < cs) ? f2b(src[(size_t)m * cs + c]) : (u16)0;
}
__global__ void k_concat_cvt(const float* a, int c1, const float* b, int c2,
                             u16* dst, int M, int cp) {
  long long i = (long long)blockIdx.x * 256 + threadIdx.x;
  if (i >= (long long)M * cp) return;
  int m = (int)(i / cp), c = (int)(i % cp);
  float v = 0.0f;
  if (c < c1) v = a[(size_t)m * c1 + c];
  else if (c < c1 + c2) v = b[(size_t)m * c2 + (c - c1)];
  dst[i] = f2b(v);
}
__global__ void k_cvt_rows(const float* src, u16* dst, int k, int n, int kp) {
  long long i = (long long)blockIdx.x * 256 + threadIdx.x;
  if (i >= (long long)kp * n) return;
  int kk = (int)(i / n), j = (int)(i % n);
  dst[i] = (kk < k) ? f2b(src[(size_t)kk * n + j]) : (u16)0;
}
__global__ void k_tcvt(const float* src, u16* dst, int k, int n, int kp) {
  // src [n, k] row-major -> dst [kp, n] = src^T with zero K-padding rows
  long long i = (long long)blockIdx.x * 256 + threadIdx.x;
  if (i >= (long long)kp * n) return;
  int kk = (int)(i / n), j = (int)(i % n);
  dst[i] = (kk < k) ? f2b(src[(size_t)j * k + kk]) : (u16)0;
}
__global__ void k_deg(const int* col, float* deg, int E) {
  int e = blockIdx.x * 256 + threadIdx.x;
  if (e < E) atomicAdd(&deg[col[e]], 1.0f);
}
__global__ void k_dis(const float* deg, float* dis, int n) {
  int i = blockIdx.x * 256 + threadIdx.x;
  if (i >= n) return;
  float d = deg[i];
  dis[i] = (d > 0.0f) ? rsqrtf(fmaxf(d, 1e-12f)) : 0.0f;
}
__global__ void k_msg_gate(const float* __restrict__ h, const float* __restrict__ eF,
                           const int* __restrict__ row, const int* __restrict__ col,
                           const float* __restrict__ dis,
                           const float* __restrict__ gW, const float* __restrict__ gb,
                           float* __restrict__ msg, float* __restrict__ gate,
                           int E, int cout) {
  int e = blockIdx.x * 256 + threadIdx.x;
  if (e >= E) return;
  int r = row[e], cl = col[e];
  float w = dis[r] * dis[cl];
  float g = gb[0];
  for (int c = 0; c < cout; ++c) {
    float m = (h[(size_t)r * cout + c] + eF[(size_t)e * cout + c]) * w;
    msg[(size_t)e * cout + c] = m;
    g += m * gW[c];
  }
  gate[e] = g;
}
__global__ void k_segmax(const float* gate, const int* col, u32* gmax, int E) {
  int e = blockIdx.x * 256 + threadIdx.x;
  if (e < E) atomicMax(&gmax[col[e]], f2ord(gate[e]));
}
__global__ void k_gexp(const float* gate, const int* col, const u32* gmax,
                       float* ge, float* gden, int E) {
  int e = blockIdx.x * 256 + threadIdx.x;
  if (e >= E) return;
  float g = expf(gate[e] - ord2f(gmax[col[e]]));
  ge[e] = g;
  atomicAdd(&gden[col[e]], g);
}
__global__ void k_aggr(const float* __restrict__ msg, const float* __restrict__ ge,
                       const float* __restrict__ gden, const int* __restrict__ col,
                       float* __restrict__ aggr, int E, int cout) {
  long long i = (long long)blockIdx.x * 256 + threadIdx.x;
  if (i >= (long long)E * cout) return;
  int e = (int)(i / cout), c = (int)(i % cout);
  int cl = col[e];
  float alpha = ge[e] / gden[cl];
  atomicAdd(&aggr[(size_t)cl * cout + c], alpha * msg[(size_t)e * cout + c]);
}
__global__ void k_gru(const float* __restrict__ gi, const float* __restrict__ gh,
                      const float* __restrict__ h, float* __restrict__ xout,
                      int n, int c) {
  long long i = (long long)blockIdx.x * 256 + threadIdx.x;
  if (i >= (long long)n * c) return;
  int m = (int)(i / c), j = (int)(i % c);
  size_t b3 = (size_t)m * 3 * c;
  float ir = gi[b3 + j], iz = gi[b3 + c + j], in_ = gi[b3 + 2 * c + j];
  float hr = gh[b3 + j], hz = gh[b3 + c + j], hn = gh[b3 + 2 * c + j];
  float r = 1.0f / (1.0f + expf(-(ir + hr)));
  float z = 1.0f / (1.0f + expf(-(iz + hz)));
  float cand = tanhf(in_ + r * hn);
  float hv = h[(size_t)m * c + j];
  xout[(size_t)m * c + j] = (1.0f - z) * cand + z * hv;
}

// ---------------------------------------------------------------------------
extern "C" void kernel_launch(void* const* d_in, const int* in_sizes, int n_in,
                              void* d_out, int out_size, void* d_ws, size_t ws_size,
                              hipStream_t stream) {
  (void)in_sizes; (void)n_in; (void)out_size; (void)ws_size;
  const int Nn = 15500, Ee = 62000;
  auto r32 = [](int x) { return (x + 31) & ~31; };
  auto F   = [&](int i) { return (const float*)d_in[i]; };

  const float* x0  = F(0);
  const float* ea0 = F(1);
  const int*   ei  = (const int*)d_in[2];
  const int*   row = ei;
  const int*   col = ei + Ee;
  // params flattened in setup_inputs() insertion order:
  // layer l at 3 + l*58: node fc(26), edge fc(26), gW, gb, Wih, Whh, bih, bhh
  // fc: cW0,cb0,...,cW9,cb9, dW1,db1,dW2,db2,dW3,db3
  const int PB = 3, LS = 58;

  // ---- workspace bump allocator (256B aligned) ----
  char* ws = (char*)d_ws; size_t off = 0;
  auto alloc = [&](size_t bytes) -> void* {
    off = (off + 255) & ~(size_t)255;
    void* p = ws + off; off += bytes; return p;
  };
  u16*   xin     = (u16*)  alloc((size_t)Nn * 224 * 2);
  u16*   ein     = (u16*)  alloc((size_t)Ee * 224 * 2);
  float* xout    = (float*)alloc((size_t)Nn * 200 * 4);
  float* eraw    = (float*)alloc((size_t)Ee * 200 * 4);
  float* hbuf    = (float*)alloc((size_t)Nn * 200 * 4);
  u16*   hbf     = (u16*)  alloc((size_t)Nn * 224 * 2);
  u16*   convOut = (u16*)  alloc((size_t)Ee * 2016 * 2);
  u16*   d1out   = (u16*)  alloc((size_t)Ee * 64 * 2);
  u16*   d2out   = (u16*)  alloc((size_t)Ee * 32 * 2);
  u16*   wbig    = (u16*)  alloc((size_t)19 * 224 * 2000 * 2);
  float* bbig    = (float*)alloc((size_t)2000 * 4);
  u16*   w1cvt   = (u16*)  alloc((size_t)2016 * 64 * 2);
  u16*   w2cvt   = (u16*)  alloc((size_t)64 * 32 * 2);
  u16*   w3cvt   = (u16*)  alloc((size_t)32 * 200 * 2);
  u16*   wihT    = (u16*)  alloc((size_t)224 * 600 * 2);
  u16*   whhT    = (u16*)  alloc((size_t)224 * 600 * 2);
  float* aggr    = (float*)alloc((size_t)Nn * 200 * 4);
  u16*   aggrbf  = (u16*)  alloc((size_t)Nn * 224 * 2);
  float* gi      = (float*)alloc((size_t)Nn * 600 * 4);
  float* gh      = (float*)alloc((size_t)Nn * 600 * 4);
  float* msg     = (float*)alloc((size_t)Ee * 200 * 4);
  float* gate    = (float*)alloc((size_t)Ee * 4);
  float* ge      = (float*)alloc((size_t)Ee * 4);
  float* deg     = (float*)alloc((size_t)Nn * 4);
  float* dis     = (float*)alloc((size_t)Nn * 4);
  u32*   gmax    = (u32*)  alloc((size_t)Nn * 4);
  float* gden    = (float*)alloc((size_t)Nn * 4);
  u16*   gbf     = (u16*)  alloc((size_t)500 * 6208 * 2);
  u16*   mw1cvt  = (u16*)  alloc((size_t)6208 * 512 * 2);
  u16*   m1bf    = (u16*)  alloc((size_t)500 * 512 * 2);
  u16*   mw2cvt  = (u16*)  alloc((size_t)512 * 32 * 2);

  auto g1  = [](long long n) { return dim3((unsigned)((n + 255) / 256)); };
  auto gemm = [&](const u16* A, int lda, const u16* B, int ldb, const float* bias,
                  float* Cf, int ldcf, u16* Cb, int ldcb, int M, int N, int K, int relu) {
    dim3 g((N + 63) / 64, (M + 127) / 128);
    k_gemm<<<g, 256, 0, stream>>>(A, lda, B, ldb, bias, Cf, ldcf, Cb, ldcb, M, N, K, relu);
  };

  // feature_conv + 3-layer MLP, all via WMMA GEMMs
  auto feature_fc = [&](int fcBase, const u16* inBf, int M, int cin, int cinP, int cout,
                        float* outF, u16* outB, int coutP) {
    int NO = 10 * cout, NOp = r32(NO);
    ConvW cw;
    for (int i = 0; i < 10; ++i) { cw.w[i] = F(fcBase + 2 * i); cw.b[i] = F(fcBase + 2 * i + 1); }
    long long tot = (long long)19 * cinP * NO;
    k_build_wbig<<<g1(tot), 256, 0, stream>>>(cw, wbig, cin, cinP, cout);
    k_bbig<<<g1(NO), 256, 0, stream>>>(cw, bbig, cout);
    dim3 gc((NO + 63) / 64, (M + 127) / 128);
    k_convgemm<<<gc, 256, 0, stream>>>(inBf, cinP, wbig, bbig, convOut, NOp, M, NO);
    k_cvt_rows<<<g1((long long)NOp * 64), 256, 0, stream>>>(F(fcBase + 20), w1cvt, NO, 64, NOp);
    gemm(convOut, NOp, w1cvt, 64, F(fcBase + 21), nullptr, 0, d1out, 64, M, 64, NOp, 1);
    k_cvt_rows<<<g1(64 * 32), 256, 0, stream>>>(F(fcBase + 22), w2cvt, 64, 32, 64);
    gemm(d1out, 64, w2cvt, 32, F(fcBase + 23), nullptr, 0, d2out, 32, M, 32, 64, 1);
    k_cvt_rows<<<g1(32 * 200), 256, 0, stream>>>(F(fcBase + 24), w3cvt, 32, cout, 32);
    gemm(d2out, 32, w3cvt, cout, F(fcBase + 25), outF, cout, outB, coutP, M, cout, 32, 0);
  };

  const int cinA[4]  = {12, 112, 112, 212};
  const int coutA[4] = {100, 100, 200, 200};

  // initial padded bf16 inputs
  k_cvt_pad<<<g1((long long)Nn * 32), 256, 0, stream>>>(x0, xin, Nn, 12, 32);
  k_cvt_pad<<<g1((long long)Ee * 32), 256, 0, stream>>>(ea0, ein, Ee, 12, 32);

  for (int l = 0; l < 4; ++l) {
    int LB = PB + l * LS;
    int cin = cinA[l], cout = coutA[l];
    int cinP = r32(cin), coutP = r32(cout);
    // node features h, edge features e
    feature_fc(LB + 0,  xin, Nn, cin, cinP, cout, hbuf, hbf, coutP);
    feature_fc(LB + 26, ein, Ee, cin, cinP, cout, eraw, nullptr, 0);
    // symmetric GCN normalization
    k_zero_f32<<<g1(Nn), 256, 0, stream>>>(deg, Nn);
    k_deg<<<g1(Ee), 256, 0, stream>>>(col, deg, Ee);
    k_dis<<<g1(Nn), 256, 0, stream>>>(deg, dis, Nn);
    // messages + attention softmax (segment max / sum)
    k_msg_gate<<<g1(Ee), 256, 0, stream>>>(hbuf, eraw, row, col, dis,
                                           F(LB + 52), F(LB + 53), msg, gate, Ee, cout);
    k_zero_u32<<<g1(Nn), 256, 0, stream>>>(gmax, Nn);
    k_segmax<<<g1(Ee), 256, 0, stream>>>(gate, col, gmax, Ee);
    k_zero_f32<<<g1(Nn), 256, 0, stream>>>(gden, Nn);
    k_gexp<<<g1(Ee), 256, 0, stream>>>(gate, col, gmax, ge, gden, Ee);
    k_zero_f32<<<g1((long long)Nn * cout), 256, 0, stream>>>(aggr, (long long)Nn * cout);
    k_aggr<<<g1((long long)Ee * cout), 256, 0, stream>>>(msg, ge, gden, col, aggr, Ee, cout);
    // GRU cell: gi = aggr @ Wih^T + bih ; gh = h @ Whh^T + bhh
    k_cvt_pad<<<g1((long long)Nn * coutP), 256, 0, stream>>>(aggr, aggrbf, Nn, cout, coutP);
    k_tcvt<<<g1((long long)coutP * 3 * cout), 256, 0, stream>>>(F(LB + 54), wihT, cout, 3 * cout, coutP);
    gemm(aggrbf, coutP, wihT, 3 * cout, F(LB + 56), gi, 3 * cout, nullptr, 0, Nn, 3 * cout, coutP, 0);
    k_tcvt<<<g1((long long)coutP * 3 * cout), 256, 0, stream>>>(F(LB + 55), whhT, cout, 3 * cout, coutP);
    gemm(hbf, coutP, whhT, 3 * cout, F(LB + 57), gh, 3 * cout, nullptr, 0, Nn, 3 * cout, coutP, 0);
    k_gru<<<g1((long long)Nn * cout), 256, 0, stream>>>(gi, gh, hbuf, xout, Nn, cout);
    // next-layer inputs: concat with original x / edge_attr
    if (l < 3) {
      int nc = r32(cout + 12);
      k_concat_cvt<<<g1((long long)Nn * nc), 256, 0, stream>>>(xout, cout, x0, 12, xin, Nn, nc);
      k_concat_cvt<<<g1((long long)Ee * nc), 256, 0, stream>>>(eraw, cout, ea0, 12, ein, Ee, nc);
    }
  }

  // final MLP: [500, 6200] -> 512 (ReLU) -> 32
  k_cvt_pad<<<g1((long long)500 * 6208), 256, 0, stream>>>(xout, gbf, 500, 6200, 6208);
  k_cvt_rows<<<g1((long long)6208 * 512), 256, 0, stream>>>(F(PB + 4 * LS + 0), mw1cvt, 6200, 512, 6208);
  gemm(gbf, 6208, mw1cvt, 512, F(PB + 4 * LS + 1), nullptr, 0, m1bf, 512, 500, 512, 6208, 1);
  k_cvt_rows<<<g1(512 * 32), 256, 0, stream>>>(F(PB + 4 * LS + 2), mw2cvt, 512, 32, 512);
  gemm(m1bf, 512, mw2cvt, 32, F(PB + 4 * LS + 3), (float*)d_out, 32, nullptr, 0, 500, 32, 512, 0);
}
